// AttentionBlock_53919019434332
// MI455X (gfx1250) — compile-verified, matrix operand order checked by reading, not run on hardware
//
#include <hip/hip_runtime.h>
#include <hip/hip_bf16.h>

typedef __attribute__((ext_vector_type(16))) _Float16 v16h;
typedef __attribute__((ext_vector_type(8)))  float    v8f;
typedef __attribute__((ext_vector_type(4)))  unsigned u32x4;
typedef __attribute__((ext_vector_type(8)))  int      i32x8;
typedef __attribute__((ext_vector_type(4)))  int      i32x4;
typedef __attribute__((ext_vector_type(2)))  int      i32x2;

#define AS1 __attribute__((address_space(1)))
#define AS3 __attribute__((address_space(3)))

#if defined(__has_builtin)
#if __has_builtin(__builtin_amdgcn_global_load_async_to_lds_b64) && \
    __has_builtin(__builtin_amdgcn_global_load_async_to_lds_b128) && \
    __has_builtin(__builtin_amdgcn_s_wait_asynccnt)
#define HAVE_ASYNC_LDS 1
#endif
#if __has_builtin(__builtin_amdgcn_tensor_load_to_lds) && \
    __has_builtin(__builtin_amdgcn_s_wait_tensorcnt)
#define HAVE_TDM 1
#endif
#endif

static constexpr int B_  = 8;
static constexpr int N_  = 1024;
static constexpr int C_  = 2048;
static constexpr int D_  = 2048;
static constexpr int BD_ = 512;

// ---------------------------------------------------------------------------
// WMMA helpers (CDNA5 wave32, 16x16x32 f16 -> f32)
// ---------------------------------------------------------------------------
__device__ __forceinline__ v16h load_frag16(const _Float16* rowp, int g) {
  // rowp points at (row, k0) of a row-major f16 matrix; lane's row already
  // selected by caller (lane % 16). Per ISA: pair r holds K = 16*(r/4) +
  // 8*g + 2*(r%4) (+0/+1)  ->  u32 index = 8*(r/4) + 4*g + (r%4).
  union { v16h h; unsigned w[8]; } f;
  const unsigned* u = (const unsigned*)rowp;
#pragma unroll
  for (int r = 0; r < 8; ++r) {
    int pi = ((r >> 2) << 3) + (g << 2) + (r & 3);
    f.w[r] = u[pi];
  }
  return f.h;
}

__device__ __forceinline__ v8f wmma16(v16h a, v16h b, v8f c) {
  return __builtin_amdgcn_wmma_f32_16x16x32_f16(false, a, false, b, (short)0, c,
                                                false, false);
}

// ---------------------------------------------------------------------------
// Prep kernels: BN scale/shift precompute + weight transpose/convert to f16
// ---------------------------------------------------------------------------
__global__ void bnprep_kernel(const float* __restrict__ gamma,
                              const float* __restrict__ beta,
                              const float* __restrict__ mean,
                              const float* __restrict__ var,
                              float* __restrict__ scale,
                              float* __restrict__ shift, int n) {
  int i = blockIdx.x * blockDim.x + threadIdx.x;
  if (i < n) {
    float s = gamma[i] * rsqrtf(var[i] + 1e-3f);
    scale[i] = s;
    shift[i] = beta[i] - mean[i] * s;
  }
}

__global__ void wtrans_kernel(const float* __restrict__ W,  // [din, dout]
                              _Float16* __restrict__ Wt,    // [dout, din]
                              int din, int dout) {
  long idx = (long)blockIdx.x * blockDim.x + threadIdx.x;
  long total = (long)din * dout;
  if (idx < total) {
    int o = (int)(idx % dout);
    int k = (int)(idx / dout);
    Wt[(long)o * din + k] = (_Float16)W[(long)k * dout + o];
  }
}

// ---------------------------------------------------------------------------
// Fused BN -> GEMM(f16 WMMA) -> +bias -> relu6 -> L2-normalize [-> row mask]
//   x  : [M, 2048] f32        Wt : [512, 2048] f16 (pre-transposed)
//   out: [M, 512]  f16
// Block = 16 rows x 512 cols; 8 waves, each wave owns 4 16x16 output tiles.
// ---------------------------------------------------------------------------
__global__ __launch_bounds__(256) void proj_qkv_kernel(
    const float* __restrict__ x, const _Float16* __restrict__ Wt,
    const float* __restrict__ scale, const float* __restrict__ shiftv,
    const float* __restrict__ bias, const unsigned char* __restrict__ vmaskp,
    _Float16* __restrict__ out, int M) {
  __shared__ _Float16 As[16 * 40];   // 16x32 tile, padded stride 40 halves
  __shared__ float rowsum[16];

  const int tid  = threadIdx.x;
  const int wave = tid >> 5;
  const int lane = tid & 31;
  const int g    = lane >> 4;
  const int cl   = lane & 15;
  const int row0 = blockIdx.x * 16;

  v8f acc[4] = {};

  for (int k0 = 0; k0 < D_; k0 += 32) {
    // Stage A tile with fused BatchNorm, f32 -> f16 (conversion => manual).
    int e  = tid * 2;
    int ar = e >> 5;
    int ac = e & 31;
    const float* xp = x + (long)(row0 + ar) * D_ + k0 + ac;
    __builtin_prefetch(xp + 32, 0, 0);
    float h0 = xp[0] * scale[k0 + ac]     + shiftv[k0 + ac];
    float h1 = xp[1] * scale[k0 + ac + 1] + shiftv[k0 + ac + 1];
    As[ar * 40 + ac]     = (_Float16)h0;
    As[ar * 40 + ac + 1] = (_Float16)h1;
    __syncthreads();

    v16h a = load_frag16(&As[cl * 40], g);
#pragma unroll
    for (int t = 0; t < 4; ++t) {
      int n0 = wave * 64 + t * 16;
      v16h b = load_frag16(Wt + (long)(n0 + cl) * D_ + k0, g);
      acc[t] = wmma16(a, b, acc[t]);
    }
    __syncthreads();
  }

  // Epilogue: bias + relu6, row-wise L2 norm across all 512 cols, store f16.
  if (tid < 16) rowsum[tid] = 0.0f;
  __syncthreads();

  float hv[4][8];
  float psum[8];
#pragma unroll
  for (int r = 0; r < 8; ++r) psum[r] = 0.0f;
#pragma unroll
  for (int t = 0; t < 4; ++t) {
    int col = wave * 64 + t * 16 + cl;
    float bb = bias[col];
#pragma unroll
    for (int r = 0; r < 8; ++r) {
      float h = acc[t][r] + bb;
      h = fminf(fmaxf(h, 0.0f), 6.0f);
      hv[t][r] = h;
      psum[r] += h * h;
    }
  }
  // Reduce over the 16 lanes of each half-wave (same rows, different cols).
#pragma unroll
  for (int r = 0; r < 8; ++r) {
    psum[r] += __shfl_xor(psum[r], 1, 32);
    psum[r] += __shfl_xor(psum[r], 2, 32);
    psum[r] += __shfl_xor(psum[r], 4, 32);
    psum[r] += __shfl_xor(psum[r], 8, 32);
  }
  if (cl == 0) {
#pragma unroll
    for (int r = 0; r < 8; ++r) atomicAdd(&rowsum[r + 8 * g], psum[r]);
  }
  __syncthreads();

#pragma unroll
  for (int r = 0; r < 8; ++r) {
    int row = row0 + r + 8 * g;
    float inv = rsqrtf(fmaxf(rowsum[r + 8 * g], 1e-12f));
    float vm  = vmaskp ? (vmaskp[row] ? 1.0f : 0.0f) : 1.0f;
    float s = inv * vm;
#pragma unroll
    for (int t = 0; t < 4; ++t) {
      int col = wave * 64 + t * 16 + cl;
      out[(long)row * BD_ + col] = (_Float16)(hv[t][r] * s);
    }
  }
}

// ---------------------------------------------------------------------------
// Streaming (flash-style) attention over 32-key chunks + fused f-proj BN.
//   Qh/Kh/Vh : [B*rows, 512] f16 (L2-normalized; V masked)
//   featbn   : [B*N, 512] f16  — BN(f) already applied, ready for final GEMM
// One block = one (b, 16-query) tile; 8 waves each own 64 output columns.
// Q tile is brought in by the Tensor Data Mover (one descriptor, wave 0),
// using TDM's LDS pad feature to produce the 520-half padded row stride.
// ---------------------------------------------------------------------------
__global__ __launch_bounds__(256) void attn_kernel(
    const _Float16* __restrict__ Qh, const _Float16* __restrict__ Kh,
    const _Float16* __restrict__ Vh, const unsigned char* __restrict__ vmaskp,
    const float* __restrict__ fscale, const float* __restrict__ fshift,
    _Float16* __restrict__ featbn) {
  __shared__ _Float16 Qs[16 * 520];   // 16 x 512, padded stride
  __shared__ float    Ss[16 * 32];    // score chunk
  __shared__ _Float16 Ps[16 * 40];    // probability chunk (f16), padded
  __shared__ _Float16 Vt[512 * 36];   // V chunk transposed [col][key], padded
  __shared__ float mrow[16], lrow[16], resc[16];

  const int tid  = threadIdx.x;
  const int wave = tid >> 5;
  const int lane = tid & 31;
  const int g    = lane >> 4;
  const int cl   = lane & 15;

  const int b    = blockIdx.x >> 6;          // N_/16 = 64 tiles per batch
  const int qb   = blockIdx.x & 63;
  const int qr0  = b * N_ + qb * 16;         // global query row

#if defined(HAVE_TDM)
  // Tensor Data Mover: 2D tile, 512 x 16 halves, data_size=2B.
  // LDS padding: pad_interval code 7 (256 DWORDs = one row) + pad_amount
  // code 3 (4 DWORDs = 8 halves) => row stride 520 halves in LDS.
  if (wave == 0) {
    unsigned ldsoff = (unsigned)(unsigned long long)(AS3 _Float16*)&Qs[0];
    unsigned long long ga =
        (unsigned long long)(size_t)(const void*)(Qh + (long)qr0 * BD_);
    u32x4 g0;
    g0[0] = 1u;                                    // count=1, user descriptor
    g0[1] = ldsoff;                                // lds_addr
    g0[2] = (unsigned)(ga & 0xFFFFFFFFu);          // global_addr[31:0]
    g0[3] = (unsigned)((ga >> 32) & 0x1FFFFFFu) | (2u << 30);  // type=2
    i32x8 g1;
    g1[0] = (1 << 16) | (1 << 20) | (7 << 22) | (3 << 25);  // dsz=2B, pad
    g1[1] = (int)(512u << 16);      // tensor_dim0[15:0] in high half
    g1[2] = (int)(16u << 16);       // tensor_dim0[31:16]=0 | tensor_dim1 lo
    g1[3] = (int)(512u << 16);      // tensor_dim1 hi=0 | tile_dim0=512
    g1[4] = 16;                     // tile_dim1=16, tile_dim2=0
    g1[5] = 512;                    // tensor_dim0_stride[31:0]
    g1[6] = 0;
    g1[7] = 0;
    i32x4 gz = {0, 0, 0, 0};
#if __clang_major__ >= 23
    i32x8 gz8 = {0, 0, 0, 0, 0, 0, 0, 0};
    __builtin_amdgcn_tensor_load_to_lds(g0, g1, gz, gz, gz8, 0);
#else
    __builtin_amdgcn_tensor_load_to_lds(g0, g1, gz, gz, 0);
#endif
    __builtin_amdgcn_s_wait_tensorcnt(0);
  }
#elif defined(HAVE_ASYNC_LDS)
  // Async copy engine: 1024 x b128, 4 per thread.
#pragma unroll
  for (int i = 0; i < 4; ++i) {
    int idx = tid + i * 256;
    int qr  = idx >> 6;                  // 64 x 16B per row
    int c8  = idx & 63;
    __builtin_amdgcn_global_load_async_to_lds_b128(
        (AS1 i32x4*)(Qh + (long)(qr0 + qr) * BD_ + c8 * 8),
        (AS3 i32x4*)&Qs[qr * 520 + c8 * 8], 0, 0);
  }
  __builtin_amdgcn_s_wait_asynccnt(0);
#else
#pragma unroll
  for (int i = 0; i < 16; ++i) {
    int idx = tid + i * 256;                 // 4096 u32 total
    int qr  = idx >> 8;
    int c2  = idx & 255;
    unsigned v = *(const unsigned*)(Qh + (long)(qr0 + qr) * BD_ + c2 * 2);
    *(unsigned*)&Qs[qr * 520 + c2 * 2] = v;
  }
#endif
  if (tid < 16) { mrow[tid] = -1e30f; lrow[tid] = 0.0f; }

  v8f acc[4] = {};
  __syncthreads();

  for (int kc = 0; kc < C_; kc += 32) {
    // (a) Stage V chunk transposed: Vt[col][key] (all threads).
#pragma unroll
    for (int i = 0; i < 32; ++i) {
      int idx = tid + i * 256;               // 8192 u32 total
      int key = idx >> 8;
      int c2  = idx & 255;
      union { unsigned u; _Float16 h[2]; } v;
      v.u = *(const unsigned*)(Vh + (long)(b * C_ + kc + key) * BD_ + c2 * 2);
      Vt[(c2 * 2)     * 36 + key] = v.h[0];
      Vt[(c2 * 2 + 1) * 36 + key] = v.h[1];
    }

    // (b) S = 100 * Q @ K^T for this chunk (waves 0,1: one 16x16 tile each).
    if (wave < 2) {
      int key0 = kc + wave * 16;
      v8f s = {};
#pragma unroll
      for (int kk = 0; kk < BD_; kk += 32) {
        v16h a = load_frag16(&Qs[cl * 520 + kk], g);
        v16h kb = load_frag16(Kh + (long)(b * C_ + key0 + cl) * BD_ + kk, g);
        s = wmma16(a, kb, s);
      }
#pragma unroll
      for (int r = 0; r < 8; ++r)
        Ss[(r + 8 * g) * 32 + wave * 16 + cl] = s[r] * 100.0f;  // 1/TEMP
    }
    __syncthreads();

    // (c) Online softmax update (one thread per query row).
    if (tid < 16) {
      float mo = mrow[tid];
      float mc = mo;
      float sv[32];
#pragma unroll
      for (int j = 0; j < 32; ++j) {
        float s = Ss[tid * 32 + j];
        if (!vmaskp[b * C_ + kc + j]) s += -1.0e7f;  // NEG_PAD / TEMP
        sv[j] = s;
        mc = fmaxf(mc, s);
      }
      float rs = __expf(mo - mc);
      float ls = 0.0f;
#pragma unroll
      for (int j = 0; j < 32; ++j) {
        float p = __expf(sv[j] - mc);
        ls += p;
        Ps[tid * 40 + j] = (_Float16)p;
      }
      mrow[tid] = mc;
      lrow[tid] = lrow[tid] * rs + ls;
      resc[tid] = rs;
    }
    __syncthreads();

    // (d) O = O*rescale + P @ V_chunk (all 8 waves, 4 tiles each).
    v16h pa = load_frag16(&Ps[cl * 40], g);
    float rsc[8];
#pragma unroll
    for (int r = 0; r < 8; ++r) rsc[r] = resc[r + 8 * g];
#pragma unroll
    for (int t = 0; t < 4; ++t) {
      int vc = wave * 64 + t * 16 + cl;
      v16h vb = load_frag16(&Vt[vc * 36], g);
#pragma unroll
      for (int r = 0; r < 8; ++r) acc[t][r] *= rsc[r];
      acc[t] = wmma16(pa, vb, acc[t]);
    }
    __syncthreads();
  }

  // Finalize: O /= l, then fuse f-projection BatchNorm, store f16.
  float linv[8];
#pragma unroll
  for (int r = 0; r < 8; ++r) linv[r] = 1.0f / lrow[r + 8 * g];
#pragma unroll
  for (int t = 0; t < 4; ++t) {
    int col = wave * 64 + t * 16 + cl;
    float sc = fscale[col];
    float sh = fshift[col];
#pragma unroll
    for (int r = 0; r < 8; ++r) {
      float f = acc[t][r] * linv[r];
      featbn[(long)(qr0 + r + 8 * g) * BD_ + col] = (_Float16)(f * sc + sh);
    }
  }
}

// ---------------------------------------------------------------------------
// Final projection: out = clip(featbn @ Wf + fb, 0, 6), f32 output [8192,2048]
// Block = 16 rows x 2048 cols; 8 waves x 16 tiles; K = 512.
// A tile staged via async global->LDS copies when available.
// ---------------------------------------------------------------------------
__global__ __launch_bounds__(256) void proj_f_kernel(
    const _Float16* __restrict__ inp,   // [M, 512] f16 (BN already applied)
    const _Float16* __restrict__ Wt,    // [2048, 512] f16
    const float* __restrict__ bias,     // [2048]
    float* __restrict__ out, int M) {
  __shared__ _Float16 As[16 * 40];

  const int tid  = threadIdx.x;
  const int wave = tid >> 5;
  const int lane = tid & 31;
  const int g    = lane >> 4;
  const int cl   = lane & 15;
  const int row0 = blockIdx.x * 16;

  v8f acc[16] = {};

  for (int k0 = 0; k0 < BD_; k0 += 32) {
#if defined(HAVE_ASYNC_LDS)
    if (tid < 128) {
      int e  = tid * 4;
      int ar = e >> 5;
      int ac = e & 31;
      __builtin_amdgcn_global_load_async_to_lds_b64(
          (AS1 i32x2*)(inp + (long)(row0 + ar) * BD_ + k0 + ac),
          (AS3 i32x2*)&As[ar * 40 + ac], 0, 0);
    }
    __builtin_amdgcn_s_wait_asynccnt(0);
#else
    int e  = tid * 2;
    int ar = e >> 5;
    int ac = e & 31;
    unsigned v = *(const unsigned*)(inp + (long)(row0 + ar) * BD_ + k0 + ac);
    *(unsigned*)&As[ar * 40 + ac] = v;
#endif
    __syncthreads();

    v16h a = load_frag16(&As[cl * 40], g);
#pragma unroll
    for (int t = 0; t < 16; ++t) {
      int n0 = wave * 256 + t * 16;
      v16h b = load_frag16(Wt + (long)(n0 + cl) * BD_ + k0, g);
      acc[t] = wmma16(a, b, acc[t]);
    }
    __syncthreads();
  }

#pragma unroll
  for (int t = 0; t < 16; ++t) {
    int col = wave * 256 + t * 16 + cl;
    float bb = bias[col];
#pragma unroll
    for (int r = 0; r < 8; ++r) {
      float h = fminf(fmaxf(acc[t][r] + bb, 0.0f), 6.0f);
      out[(long)(row0 + r + 8 * g) * D_ + col] = h;
    }
  }
}

// ---------------------------------------------------------------------------
extern "C" void kernel_launch(void* const* d_in, const int* in_sizes, int n_in,
                              void* d_out, int out_size, void* d_ws,
                              size_t ws_size, hipStream_t stream) {
  const float* x_in  = (const float*)d_in[0];
  const float* ctx   = (const float*)d_in[1];
  const unsigned char* vmask = (const unsigned char*)d_in[2];
  const float *q_gamma = (const float*)d_in[3],  *q_beta = (const float*)d_in[4],
              *q_mean  = (const float*)d_in[5],  *q_var  = (const float*)d_in[6],
              *q_W     = (const float*)d_in[7],  *q_b    = (const float*)d_in[8];
  const float *k_gamma = (const float*)d_in[9],  *k_beta = (const float*)d_in[10],
              *k_mean  = (const float*)d_in[11], *k_var  = (const float*)d_in[12],
              *k_W     = (const float*)d_in[13], *k_b    = (const float*)d_in[14];
  const float *v_gamma = (const float*)d_in[15], *v_beta = (const float*)d_in[16],
              *v_mean  = (const float*)d_in[17], *v_var  = (const float*)d_in[18],
              *v_W     = (const float*)d_in[19], *v_b    = (const float*)d_in[20];
  const float *f_gamma = (const float*)d_in[21], *f_beta = (const float*)d_in[22],
              *f_mean  = (const float*)d_in[23], *f_var  = (const float*)d_in[24],
              *f_W     = (const float*)d_in[25], *f_b    = (const float*)d_in[26];

  char* ws = (char*)d_ws;
  size_t off = 0;
  _Float16* Wq_t = (_Float16*)(ws + off); off += (size_t)BD_ * D_ * 2;
  _Float16* Wk_t = (_Float16*)(ws + off); off += (size_t)BD_ * D_ * 2;
  _Float16* Wv_t = (_Float16*)(ws + off); off += (size_t)BD_ * D_ * 2;
  _Float16* Wf_t = (_Float16*)(ws + off); off += (size_t)D_ * BD_ * 2;
  _Float16* Qh   = (_Float16*)(ws + off); off += (size_t)B_ * N_ * BD_ * 2;
  _Float16* Kh   = (_Float16*)(ws + off); off += (size_t)B_ * C_ * BD_ * 2;
  _Float16* Vh   = (_Float16*)(ws + off); off += (size_t)B_ * C_ * BD_ * 2;
  _Float16* Fbn  = (_Float16*)(ws + off); off += (size_t)B_ * N_ * BD_ * 2;
  float* qsc = (float*)(ws + off); off += D_ * 4;
  float* qsh = (float*)(ws + off); off += D_ * 4;
  float* ksc = (float*)(ws + off); off += D_ * 4;
  float* ksh = (float*)(ws + off); off += D_ * 4;
  float* vsc = (float*)(ws + off); off += D_ * 4;
  float* vsh = (float*)(ws + off); off += D_ * 4;
  float* fsc = (float*)(ws + off); off += BD_ * 4;
  float* fsh = (float*)(ws + off); off += BD_ * 4;

  // 1) BN scale/shift precompute.
  bnprep_kernel<<<(D_ + 255) / 256, 256, 0, stream>>>(q_gamma, q_beta, q_mean, q_var, qsc, qsh, D_);
  bnprep_kernel<<<(D_ + 255) / 256, 256, 0, stream>>>(k_gamma, k_beta, k_mean, k_var, ksc, ksh, D_);
  bnprep_kernel<<<(D_ + 255) / 256, 256, 0, stream>>>(v_gamma, v_beta, v_mean, v_var, vsc, vsh, D_);
  bnprep_kernel<<<(BD_ + 255) / 256, 256, 0, stream>>>(f_gamma, f_beta, f_mean, f_var, fsc, fsh, BD_);

  // 2) Weight transpose + f16 conversion.
  long wqe = (long)D_ * BD_;
  wtrans_kernel<<<(unsigned)((wqe + 255) / 256), 256, 0, stream>>>(q_W, Wq_t, D_, BD_);
  wtrans_kernel<<<(unsigned)((wqe + 255) / 256), 256, 0, stream>>>(k_W, Wk_t, D_, BD_);
  wtrans_kernel<<<(unsigned)((wqe + 255) / 256), 256, 0, stream>>>(v_W, Wv_t, D_, BD_);
  wtrans_kernel<<<(unsigned)((wqe + 255) / 256), 256, 0, stream>>>(f_W, Wf_t, BD_, D_);

  // 3) Fused BN+GEMM+relu6+L2norm projections (WMMA).
  proj_qkv_kernel<<<(B_ * N_) / 16, 256, 0, stream>>>(x_in, Wq_t, qsc, qsh, q_b, nullptr, Qh, B_ * N_);
  proj_qkv_kernel<<<(B_ * C_) / 16, 256, 0, stream>>>(ctx,  Wk_t, ksc, ksh, k_b, nullptr, Kh, B_ * C_);
  proj_qkv_kernel<<<(B_ * C_) / 16, 256, 0, stream>>>(ctx,  Wv_t, vsc, vsh, v_b, vmask,  Vh, B_ * C_);

  // 4) Streaming attention (TDM Q-tile load) with fused f-projection BN.
  attn_kernel<<<B_ * (N_ / 16), 256, 0, stream>>>(Qh, Kh, Vh, vmask, fsc, fsh, Fbn);

  // 5) Final projection to f32 output (WMMA, async LDS staging).
  proj_f_kernel<<<(B_ * N_) / 16, 256, 0, stream>>>(Fbn, Wf_t, f_b, (float*)d_out, B_ * N_);
}
